// CausalLTXAttention_23390391894836
// MI455X (gfx1250) — matrix operand order repaired and checked by reading, not compiled, not run on hardware
//
#include <hip/hip_runtime.h>
#include <hip/hip_bf16.h>

// ---------------------------------------------------------------------------
// CDNA5 (gfx1250) causal attention block:  QKV proj -> rmsnorm+rope -> flash
// attention -> output proj, all matmuls on v_wmma_f32_16x16x32_bf16.
// Round 2: pre-transposed bf16 weights, V produced pre-transposed, and
// double-buffered LDS tiles (one barrier per K-step) in GEMM + attention.
// ---------------------------------------------------------------------------

typedef __attribute__((ext_vector_type(16))) __bf16 v16bf;
typedef __attribute__((ext_vector_type(8)))  float  v8f;

static constexpr int Bsz   = 2;
static constexpr int Lseq  = 2048;
static constexpr int Dmod  = 2048;
static constexpr int HEADS = 16;
static constexpr int DHEAD = 128;
static constexpr int INNER = HEADS * DHEAD;   // 2048
static constexpr int MROWS = Bsz * Lseq;      // 4096

#define WMMA_BF16(a, b, c) \
  __builtin_amdgcn_wmma_f32_16x16x32_bf16(false, (a), false, (b), (short)0, (c), false, false)

// 16 contiguous bf16 (B-fragment: n = lane&15 column, k = 16*(lane>>4)+e)
static __device__ inline v16bf load16(const __bf16* p) {
  union { uint4 u[2]; v16bf v; } r;
  r.u[0] = *(const uint4*)(p);
  r.u[1] = *(const uint4*)(p + 8);
  return r.v;
}
// A-fragment: row = lane&15, 16 contiguous elems at k = 16*kh.
// Call with p = rowBase + 16*kh.
static __device__ inline v16bf load_a(const __bf16* p) {
  union { uint4 u[2]; v16bf v; } r;
  r.u[0] = *(const uint4*)(p);
  r.u[1] = *(const uint4*)(p + 8);
  return r.v;
}

static __device__ inline v8f v8f_zero() {
  v8f z = {0.f, 0.f, 0.f, 0.f, 0.f, 0.f, 0.f, 0.f};
  return z;
}

// ---------------------------------------------------------------------------
// Kernel 0: f32 -> bf16 conversion
// ---------------------------------------------------------------------------
__global__ __launch_bounds__(256) void k_f32_to_bf16(const float* __restrict__ in,
                                                     __bf16* __restrict__ out, int n) {
  int i = blockIdx.x * 256 + threadIdx.x;
  if (i < n) out[i] = (__bf16)in[i];
}

// ---------------------------------------------------------------------------
// Kernel 0b: one-time W[K][N] (f32) -> Wt[N][K] (bf16), LDS-tiled 32x32 so
// both global read and global write are coalesced.
// ---------------------------------------------------------------------------
__global__ __launch_bounds__(256) void k_transpose_w(const float* __restrict__ W,
                                                     __bf16* __restrict__ Wt,
                                                     int K, int N) {
  __shared__ __bf16 t[32][33];
  const int n0 = blockIdx.x * 32;
  const int k0 = blockIdx.y * 32;
  const int tx = threadIdx.x & 31;
  const int ty = threadIdx.x >> 5;   // 0..7
#pragma unroll
  for (int i = 0; i < 32; i += 8)
    t[ty + i][tx] = (__bf16)W[(size_t)(k0 + ty + i) * N + n0 + tx];
  __syncthreads();
#pragma unroll
  for (int i = 0; i < 32; i += 8)
    Wt[(size_t)(n0 + ty + i) * K + k0 + tx] = t[tx][ty + i];
}

// ---------------------------------------------------------------------------
// Kernel 1/4: C = A(bf16 [M][K]) @ Wt(bf16 [N][K])^T + bias.  128x128 block
// tile, 8 waves each 32x64 (2x4 WMMA accums), K-step 32, double-buffered LDS.
// OUT_MODE: 0 = bf16 row-major, 1 = f32 row-major, 2 = bf16 V-transposed
// ([b][h][d][L] layout for the attention kernel's V^T staging).
// ---------------------------------------------------------------------------
template <int OUT_MODE>
__global__ __launch_bounds__(256) void k_gemm(const __bf16* __restrict__ A,
                                              const __bf16* __restrict__ Wt,
                                              const float* __restrict__ bias,
                                              void* __restrict__ outp,
                                              int M, int K, int N) {
  constexpr int AS = 40;  // padded row stride (bf16 elems), keeps 16B alignment
  __shared__ __align__(16) __bf16 At[2][128 * AS];  // A tile  [m][k]
  __shared__ __align__(16) __bf16 Bt[2][128 * AS];  // W tile  [n][k]

  const int tid  = threadIdx.x;
  const int lane = tid & 31;
  const int wave = tid >> 5;
  const int wm   = wave >> 1;      // 0..3  (m direction, 32 rows each)
  const int wn   = wave & 1;       // 0..1  (n direction, 64 cols each)
  const int m0   = blockIdx.y * 128;
  const int n0   = blockIdx.x * 128;
  const int lm   = lane & 15;
  const int kh   = lane >> 4;

  v8f acc[2][4];
#pragma unroll
  for (int i = 0; i < 2; ++i)
#pragma unroll
    for (int j = 0; j < 4; ++j) acc[i][j] = v8f_zero();

  // both A and W tiles are [128 rows][32 k] of bf16: 32B per thread per tile
  const int srow = tid >> 1;
  const int sch  = (tid & 1) * 16;

  auto stage = [&](int p, int k0) {
    const __bf16* ga = A  + (size_t)(m0 + srow) * K + k0 + sch;
    const __bf16* gw = Wt + (size_t)(n0 + srow) * K + k0 + sch;
    __builtin_prefetch(ga + 64, 0, 0);      // global_prefetch two tiles ahead
    uint4 a0 = *(const uint4*)(ga);
    uint4 a1 = *(const uint4*)(ga + 8);
    uint4 w0 = *(const uint4*)(gw);
    uint4 w1 = *(const uint4*)(gw + 8);
    *(uint4*)&At[p][srow * AS + sch]     = a0;
    *(uint4*)&At[p][srow * AS + sch + 8] = a1;
    *(uint4*)&Bt[p][srow * AS + sch]     = w0;
    *(uint4*)&Bt[p][srow * AS + sch + 8] = w1;
  };

  stage(0, 0);
  int p = 0;
  for (int k0 = 0; k0 < K; k0 += 32) {
    __syncthreads();                       // buffer p fully staged
    if (k0 + 32 < K) stage(p ^ 1, k0 + 32);

    v16bf afrag[2];
#pragma unroll
    for (int mf = 0; mf < 2; ++mf)
      afrag[mf] = load_a(&At[p][(wm * 32 + mf * 16 + lm) * AS + 16 * kh]);

#pragma unroll
    for (int nf = 0; nf < 4; ++nf) {
      v16bf bfrag = load16(&Bt[p][(wn * 64 + nf * 16 + lm) * AS + 16 * kh]);
      acc[0][nf] = WMMA_BF16(afrag[0], bfrag, acc[0][nf]);
      acc[1][nf] = WMMA_BF16(afrag[1], bfrag, acc[1][nf]);
    }
    p ^= 1;
  }

  // epilogue: C layout m = v + 8*kh, n = lane&15
#pragma unroll
  for (int mf = 0; mf < 2; ++mf)
#pragma unroll
    for (int nf = 0; nf < 4; ++nf)
#pragma unroll
      for (int v = 0; v < 8; ++v) {
        const int m = m0 + wm * 32 + mf * 16 + v + 8 * kh;
        const int n = n0 + wn * 64 + nf * 16 + lm;
        const float val = acc[mf][nf][v] + bias[n];
        if (OUT_MODE == 1) {
          ((float*)outp)[(size_t)m * N + n] = val;
        } else if (OUT_MODE == 2) {
          const int bb = m / Lseq, l  = m % Lseq;
          const int hh = n / DHEAD, dl = n % DHEAD;
          ((__bf16*)outp)[(((size_t)bb * HEADS + hh) * DHEAD + dl) * Lseq + l] =
              (__bf16)val;
        } else {
          ((__bf16*)outp)[(size_t)m * N + n] = (__bf16)val;
        }
      }
}

// ---------------------------------------------------------------------------
// Kernel 2: in-place RMSNorm + interleaved RoPE on bf16 q / k rows.
// q additionally gets logit_log_scale[l] * 1/sqrt(DHEAD) folded in.
// ---------------------------------------------------------------------------
__global__ __launch_bounds__(256) void k_norm_rope(__bf16* __restrict__ qb,
                                                   __bf16* __restrict__ kb,
                                                   const float* __restrict__ qn_w,
                                                   const float* __restrict__ kn_w,
                                                   const float* __restrict__ pe_cos,
                                                   const float* __restrict__ pe_sin,
                                                   const float* __restrict__ lls) {
  const int row   = blockIdx.x;        // 0 .. B*L-1
  const int which = blockIdx.y;        // 0 = q, 1 = k
  const int l     = row % Lseq;
  __bf16* buf     = (which == 0 ? qb : kb) + (size_t)row * INNER;
  const float* w  = (which == 0 ? qn_w : kn_w);
  const int t     = threadIdx.x;

  float x[8];
#pragma unroll
  for (int j = 0; j < 8; ++j) x[j] = (float)buf[t * 8 + j];
  float ss = 0.f;
#pragma unroll
  for (int j = 0; j < 8; ++j) ss += x[j] * x[j];

  __shared__ float red[256];
  red[t] = ss;
  __syncthreads();
  for (int s = 128; s > 0; s >>= 1) {
    if (t < s) red[t] += red[t + s];
    __syncthreads();
  }
  const float inv = rsqrtf(red[0] / (float)INNER + 1e-6f);
  const float extra = (which == 0) ? lls[l] * 0.08838834764831845f /* 1/sqrt(128) */
                                   : 1.0f;
  const float* cr = pe_cos + (size_t)l * (INNER / 2);
  const float* sr = pe_sin + (size_t)l * (INNER / 2);
#pragma unroll
  for (int p = 0; p < 4; ++p) {
    const int pi = t * 4 + p;
    const float c = cr[pi], s = sr[pi];
    const float x1 = x[2 * p]     * inv * w[t * 8 + 2 * p];
    const float x2 = x[2 * p + 1] * inv * w[t * 8 + 2 * p + 1];
    buf[t * 8 + 2 * p]     = (__bf16)((x1 * c - x2 * s) * extra);
    buf[t * 8 + 2 * p + 1] = (__bf16)((x1 * s + x2 * c) * extra);
  }
}

// ---------------------------------------------------------------------------
// Kernel 3: causal flash attention.  Block = 8 waves x 16 q-rows = 128 rows.
// 32-key tiles, double-buffered in LDS (K natural [key][d], V pre-transposed
// in global so staging is pure 32B vector copies).  S = Q K^T (4 bf16 WMMAs
// per 16-col half), online softmax via shfl_xor row reductions, P routed
// through per-wave LDS slab to A-fragment layout, O += P V (8 bf16 WMMAs).
// Softmax scale / logit scale are pre-folded into q.
// ---------------------------------------------------------------------------
__global__ __launch_bounds__(256) void k_attn(const __bf16* __restrict__ qb,
                                              const __bf16* __restrict__ kb,
                                              const __bf16* __restrict__ vT,
                                              __bf16* __restrict__ ab) {
  constexpr int KT = 136;  // K tile row stride  (128 + 8 pad)
  constexpr int VT = 40;   // V^T row stride     (32 + 8 pad)
  constexpr int PT = 40;   // P slab row stride  (32 + 8 pad)
  __shared__ __align__(16) __bf16 Kt[2][32 * KT];
  __shared__ __align__(16) __bf16 Vt[2][128 * VT];
  __shared__ __align__(16) __bf16 Ps[8 * 16 * PT];

  const int tid  = threadIdx.x;
  const int lane = tid & 31;
  const int wave = tid >> 5;
  const int lm   = lane & 15;
  const int kh   = lane >> 4;
  const int bh   = blockIdx.y;
  const int b    = bh / HEADS;
  const int h    = bh % HEADS;
  const int qb0  = blockIdx.x * 128;
  const int qrow0 = qb0 + wave * 16;

  // Q fragments (A layout), reused across the whole key loop.
  const __bf16* qrow = qb + ((size_t)(b * Lseq + qrow0 + lm) * INNER + h * DHEAD);
  v16bf qf[4];
#pragma unroll
  for (int c = 0; c < 4; ++c) qf[c] = load_a(qrow + c * 32 + 16 * kh);

  v8f O[8];
#pragma unroll
  for (int g = 0; g < 8; ++g) O[g] = v8f_zero();
  float mrun[8], lrun[8];
#pragma unroll
  for (int v = 0; v < 8; ++v) { mrun[v] = -1e30f; lrun[v] = 0.f; }

  // staging mappings (all 32B vector copies)
  const int kkey = tid >> 3;            // K tile: key row, 8 chunks of 16 d
  const int kch  = (tid & 7) * 16;
  const int vd   = tid >> 1;            // V tile: d row, 2 chunks of 16 keys
  const int vch  = (tid & 1) * 16;
  const __bf16* vrowg = vT + (((size_t)b * HEADS + h) * DHEAD + vd) * Lseq;

  auto stage = [&](int p, int kb0) {
    const __bf16* gk = kb + ((size_t)(b * Lseq + kb0 + kkey) * INNER + h * DHEAD + kch);
    uint4 k0 = *(const uint4*)(gk);
    uint4 k1 = *(const uint4*)(gk + 8);
    *(uint4*)&Kt[p][kkey * KT + kch]     = k0;
    *(uint4*)&Kt[p][kkey * KT + kch + 8] = k1;
    const __bf16* gv = vrowg + kb0 + vch;
    uint4 v0 = *(const uint4*)(gv);
    uint4 v1 = *(const uint4*)(gv + 8);
    *(uint4*)&Vt[p][vd * VT + vch]     = v0;
    *(uint4*)&Vt[p][vd * VT + vch + 8] = v1;
  };

  const int kend = qb0 + 128;
  stage(0, 0);
  int p = 0;
  for (int kb0 = 0; kb0 < kend; kb0 += 32) {
    __syncthreads();                       // buffer p fully staged
    if (kb0 + 32 < kend) stage(p ^ 1, kb0 + 32);

    // ---- scores: two 16-key halves
    v8f S[2];
#pragma unroll
    for (int nb = 0; nb < 2; ++nb) {
      S[nb] = v8f_zero();
#pragma unroll
      for (int c = 0; c < 4; ++c) {
        v16bf bfrag = load16(&Kt[p][(nb * 16 + lm) * KT + c * 32 + 16 * kh]);
        S[nb] = WMMA_BF16(qf[c], bfrag, S[nb]);
      }
    }

    // ---- causal mask + online softmax (row m = v + 8*kh, col n = lm)
    float p0[8], p1[8];
#pragma unroll
    for (int v = 0; v < 8; ++v) {
      const int m  = v + 8 * kh;
      const int qi = qrow0 + m;
      float s0 = (kb0 + lm      <= qi) ? S[0][v] : -1e30f;
      float s1 = (kb0 + 16 + lm <= qi) ? S[1][v] : -1e30f;
      float mx = fmaxf(s0, s1);
      mx = fmaxf(mx, __shfl_xor(mx, 1));
      mx = fmaxf(mx, __shfl_xor(mx, 2));
      mx = fmaxf(mx, __shfl_xor(mx, 4));
      mx = fmaxf(mx, __shfl_xor(mx, 8));
      const float mnew  = fmaxf(mrun[v], mx);
      const float alpha = __expf(mrun[v] - mnew);
      const float e0 = __expf(s0 - mnew);
      const float e1 = __expf(s1 - mnew);
      float rs = e0 + e1;
      rs += __shfl_xor(rs, 1);
      rs += __shfl_xor(rs, 2);
      rs += __shfl_xor(rs, 4);
      rs += __shfl_xor(rs, 8);
      lrun[v] = lrun[v] * alpha + rs;
      mrun[v] = mnew;
      p0[v] = e0;
      p1[v] = e1;
#pragma unroll
      for (int g = 0; g < 8; ++g) O[g][v] *= alpha;
    }

    // ---- P (C layout) -> per-wave LDS slab -> A fragment (DS ops from one
    // wave complete in order; just stop compiler reordering)
    __bf16* ps = &Ps[wave * 16 * PT];
#pragma unroll
    for (int v = 0; v < 8; ++v) {
      const int m = v + 8 * kh;
      ps[m * PT + lm]      = (__bf16)p0[v];
      ps[m * PT + 16 + lm] = (__bf16)p1[v];
    }
    asm volatile("" ::: "memory");
    v16bf pa = load_a(ps + lm * PT + 16 * kh);

    // ---- O += P V   (8 column groups of 16)
#pragma unroll
    for (int g = 0; g < 8; ++g) {
      v16bf bfrag = load16(&Vt[p][(g * 16 + lm) * VT + 16 * kh]);
      O[g] = WMMA_BF16(pa, bfrag, O[g]);
    }
    p ^= 1;
  }

  // ---- normalize and store bf16 attention output
#pragma unroll
  for (int v = 0; v < 8; ++v) {
    const float invl = 1.0f / lrun[v];
    const int m  = v + 8 * kh;
    const int qi = qrow0 + m;
    __bf16* orow = ab + ((size_t)(b * Lseq + qi) * INNER + h * DHEAD);
#pragma unroll
    for (int g = 0; g < 8; ++g)
      orow[g * 16 + lm] = (__bf16)(O[g][v] * invl);
  }
}

// ---------------------------------------------------------------------------
// Host-side launcher
// ---------------------------------------------------------------------------
extern "C" void kernel_launch(void* const* d_in, const int* in_sizes, int n_in,
                              void* d_out, int out_size, void* d_ws, size_t ws_size,
                              hipStream_t stream) {
  const float* x      = (const float*)d_in[0];
  const float* pe_cos = (const float*)d_in[1];
  const float* pe_sin = (const float*)d_in[2];
  const float* lls    = (const float*)d_in[3];
  const float* wq     = (const float*)d_in[4];
  const float* bq     = (const float*)d_in[5];
  const float* wk     = (const float*)d_in[6];
  const float* bk     = (const float*)d_in[7];
  const float* wv     = (const float*)d_in[8];
  const float* bv     = (const float*)d_in[9];
  const float* qn_w   = (const float*)d_in[10];
  const float* kn_w   = (const float*)d_in[11];
  const float* wo     = (const float*)d_in[12];
  const float* bo     = (const float*)d_in[13];
  float* out = (float*)d_out;

  const size_t NELEM = (size_t)MROWS * INNER;   // 8.39M activation elems
  const size_t WELEM = (size_t)Dmod * INNER;    // 4.19M weight elems
  __bf16* xb   = (__bf16*)d_ws;
  __bf16* qbuf = xb    + NELEM;
  __bf16* kbuf = qbuf  + NELEM;
  __bf16* vTbf = kbuf  + NELEM;   // V stored transposed: [b][h][d][L]
  __bf16* abuf = vTbf  + NELEM;
  __bf16* wqT  = abuf  + NELEM;   // weights transposed to [N][K] bf16
  __bf16* wkT  = wqT   + WELEM;
  __bf16* wvT  = wkT   + WELEM;
  __bf16* woT  = wvT   + WELEM;   // total ~117 MB of workspace

  // 0) x -> bf16 ; weights -> transposed bf16 (one-time)
  k_f32_to_bf16<<<dim3((unsigned)(NELEM / 256)), 256, 0, stream>>>(x, xb, (int)NELEM);
  dim3 tgrid(INNER / 32, Dmod / 32);   // (64, 64)
  k_transpose_w<<<tgrid, 256, 0, stream>>>(wq, wqT, Dmod, INNER);
  k_transpose_w<<<tgrid, 256, 0, stream>>>(wk, wkT, Dmod, INNER);
  k_transpose_w<<<tgrid, 256, 0, stream>>>(wv, wvT, Dmod, INNER);
  k_transpose_w<<<dim3(Dmod / 32, INNER / 32), 256, 0, stream>>>(wo, woT, INNER, Dmod);

  // 1) QKV projections (bf16 out; V written pre-transposed)
  dim3 ggrid(INNER / 128, MROWS / 128);  // (16, 32)
  k_gemm<0><<<ggrid, 256, 0, stream>>>(xb, wqT, bq, qbuf, MROWS, Dmod, INNER);
  k_gemm<0><<<ggrid, 256, 0, stream>>>(xb, wkT, bk, kbuf, MROWS, Dmod, INNER);
  k_gemm<2><<<ggrid, 256, 0, stream>>>(xb, wvT, bv, vTbf, MROWS, Dmod, INNER);

  // 2) rmsnorm + rope + scale, in place on q/k
  k_norm_rope<<<dim3(MROWS, 2), 256, 0, stream>>>(qbuf, kbuf, qn_w, kn_w,
                                                  pe_cos, pe_sin, lls);

  // 3) causal flash attention -> abuf (bf16)
  k_attn<<<dim3(Lseq / 128, Bsz * HEADS), 256, 0, stream>>>(qbuf, kbuf, vTbf, abuf);

  // 4) output projection (f32 out)
  k_gemm<1><<<dim3(Dmod / 128, MROWS / 128), 256, 0, stream>>>(abuf, woT, bo, out,
                                                               MROWS, INNER, Dmod);
}
